// DepthwiseXCorr_42030549958765
// MI455X (gfx1250) — matrix operand order repaired and checked by reading, not compiled, not run on hardware
//
#include <hip/hip_runtime.h>
#include <hip/hip_bf16.h>
#include <stdint.h>

// ---------------------------------------------------------------------------
// MI455X (gfx1250): SiamRPN depthwise-xcorr head.
//   conv towers : implicit GEMM, v_wmma_f32_16x16x32_bf16, 8 WMMA/wave/K-step,
//                 LDS im2col offset table, double-buffered A staging
//   xcorr       : VALU (1 GFLOP), LDS-staged search tile
//   head        : fused double GEMM through LDS (1x1 -> BN/ReLU -> 1x1 + bias)
// ---------------------------------------------------------------------------

typedef __bf16 bf16_t;
typedef __attribute__((ext_vector_type(16))) __bf16 v16bf;
typedef __attribute__((ext_vector_type(8)))  __bf16 v8bf;
typedef __attribute__((ext_vector_type(8)))  float  v8f;

union Bf16x16 { v16bf v; v8bf h[2]; };

__device__ __forceinline__ v8f wmma_bf16(const Bf16x16& a, const Bf16x16& b, v8f c) {
  return __builtin_amdgcn_wmma_f32_16x16x32_bf16(false, a.v, false, b.v,
                                                 (short)0, c, false, false);
}

// ------------------------------ prep kernels -------------------------------

__global__ void cvt_f32_to_bf16(const float* __restrict__ src,
                                bf16_t* __restrict__ dst, int n) {
  int i = blockIdx.x * blockDim.x + threadIdx.x;
  if (i < n) dst[i] = (bf16_t)src[i];
}

__global__ void bn_fold(const float* __restrict__ g, const float* __restrict__ b,
                        const float* __restrict__ m, const float* __restrict__ v,
                        float* __restrict__ scale, float* __restrict__ bias) {
  int i = threadIdx.x;  // 256 channels
  float s = g[i] * rsqrtf(v[i] + 1e-5f);
  scale[i] = s;
  bias[i]  = b[i] - m[i] * s;
}

// ------------------------ conv3x3 + BN + ReLU (WMMA) -----------------------
// GEMM view: C[M=B*HO*WO][N=256] = im2col(x)[M][K=2304] * Wb[N][K]^T
// Block tile 64M x 256N, 8 waves in 4(M) x 2(N) grid, 8 N-tiles per wave.
// A double-buffered in LDS; B streamed from L2 (row-major [N][K] bf16).

__global__ __launch_bounds__(256)
void conv3x3_bnrelu_wmma(const float* __restrict__ x,
                         const bf16_t* __restrict__ wb,   // [256][2304]
                         const float* __restrict__ scale,
                         const float* __restrict__ bias,
                         float* __restrict__ out,         // NCHW f32
                         int HI, int WI, int HO, int WO) {
  const int Cin = 256, K = 2304;
  __shared__ __align__(16) bf16_t As[2][64][40];
  __shared__ __align__(16) int    offTab[2304];

  const int tid   = threadIdx.x;
  const int lane  = tid & 31;
  const int wave  = tid >> 5;
  const int waveM = wave & 3;    // 4 M tiles
  const int waveN = wave >> 2;   // 2 N half-blocks of 128
  const int HW    = HO * WO;
  const long m0   = (long)blockIdx.x * 64;

  // im2col offset table (relative to pixel base): off(k)=ci*HI*WI + kh*WI + kw
  for (int k = tid; k < K; k += 256) {
    int ci = k / 9, r = k - ci * 9;
    int kh = r / 3, kw = r - kh * 3;
    offTab[k] = (ci * HI + kh) * WI + kw;
  }

  // A-staging coordinates: one m row, 8 consecutive k per thread
  const int  mloc = tid & 63;
  const int  kgrp = (tid >> 6) * 8;
  const long m    = m0 + mloc;
  const int  bImg = (int)(m / HW);
  const int  p    = (int)(m % HW);
  const int  oy   = p / WO, ox = p - oy * WO;
  const float* xpix = x + ((long)bImg * Cin * HI + oy) * WI + ox;

  __syncthreads();  // offTab ready

  // prologue: stage k0=0 into buffer 0
  {
    int4 o0 = *(const int4*)&offTab[kgrp];
    int4 o1 = *(const int4*)&offTab[kgrp + 4];
    bf16_t* dst = &As[0][mloc][kgrp];
    dst[0] = (bf16_t)xpix[o0.x];  dst[1] = (bf16_t)xpix[o0.y];
    dst[2] = (bf16_t)xpix[o0.z];  dst[3] = (bf16_t)xpix[o0.w];
    dst[4] = (bf16_t)xpix[o1.x];  dst[5] = (bf16_t)xpix[o1.y];
    dst[6] = (bf16_t)xpix[o1.z];  dst[7] = (bf16_t)xpix[o1.w];
  }

  v8f acc[8] = {};
  const int nLo  = lane & 15;
  const int bKH  = (lane < 16) ? 0 : 8;
  const int aRow = waveM * 16 + nLo;
  const int aK   = (lane < 16) ? 0 : 8;
  const bf16_t* wrow = wb + (long)(waveN * 128 + nLo) * K;

  int buf = 0;
  for (int k0 = 0; k0 < K; k0 += 32, buf ^= 1) {
    __syncthreads();
    if (k0 + 32 < K) {  // stage next slab into the other buffer
      int kk = k0 + 32 + kgrp;
      int4 o0 = *(const int4*)&offTab[kk];
      int4 o1 = *(const int4*)&offTab[kk + 4];
      bf16_t* dst = &As[buf ^ 1][mloc][kgrp];
      dst[0] = (bf16_t)xpix[o0.x];  dst[1] = (bf16_t)xpix[o0.y];
      dst[2] = (bf16_t)xpix[o0.z];  dst[3] = (bf16_t)xpix[o0.w];
      dst[4] = (bf16_t)xpix[o1.x];  dst[5] = (bf16_t)xpix[o1.y];
      dst[6] = (bf16_t)xpix[o1.z];  dst[7] = (bf16_t)xpix[o1.w];
    }

    Bf16x16 a;
    a.h[0] = *(const v8bf*)&As[buf][aRow][aK];
    a.h[1] = *(const v8bf*)&As[buf][aRow][aK + 16];
#pragma unroll
    for (int t = 0; t < 8; ++t) {
      const bf16_t* bp = wrow + (long)t * 16 * K + k0 + bKH;
      Bf16x16 bfr;
      bfr.h[0] = *(const v8bf*)(bp);
      bfr.h[1] = *(const v8bf*)(bp + 16);
      acc[t] = wmma_bf16(a, bfr, acc[t]);
    }
  }

  // Epilogue: BN + ReLU, scatter to NCHW f32
  float sc[8], bi[8];
#pragma unroll
  for (int t = 0; t < 8; ++t) {
    int n = waveN * 128 + t * 16 + nLo;
    sc[t] = scale[n];
    bi[t] = bias[n];
  }
  const int mbase = waveM * 16 + ((lane < 16) ? 0 : 8);
#pragma unroll
  for (int v = 0; v < 8; ++v) {
    long mg = m0 + mbase + v;
    int  bb = (int)(mg / HW);
    int  pp = (int)(mg % HW);
    float* orow = out + (long)bb * 256 * HW + pp;
#pragma unroll
    for (int t = 0; t < 8; ++t) {
      int n = waveN * 128 + t * 16 + nLo;
      float val = acc[t][v] * sc[t] + bi[t];
      val = val > 0.f ? val : 0.f;
      orow[(long)n * HW] = val;
    }
  }
}

// --------------------------- depthwise xcorr -------------------------------
// One wave per (b,c): 5x5 kernel over 29x29 search -> 25x25, writes bf16 NHWC.

__global__ __launch_bounds__(256)
void xcorr_dw(const float* __restrict__ sfeat,   // [B*C][841] f32
              const float* __restrict__ kfeat,   // [B*C][25]  f32
              bf16_t* __restrict__ featb) {      // [B*625][256] bf16 (NHWC)
  __shared__ float Ssh[8][848];
  const int lane = threadIdx.x & 31;
  const int wave = threadIdx.x >> 5;
  const int widx = blockIdx.x * 8 + wave;   // == b*256 + c
  const int b = widx >> 8;
  const int c = widx & 255;

  const float* sp = sfeat + (long)widx * 841;
  for (int i = lane; i < 841; i += 32) Ssh[wave][i] = sp[i];

  float kr[25];
  const float* kp = kfeat + (long)widx * 25;
#pragma unroll
  for (int i = 0; i < 25; ++i) kr[i] = kp[i];
  __syncthreads();

  bf16_t* op = featb + (long)b * 625 * 256 + c;
  for (int p = lane; p < 625; p += 32) {
    int y = p / 25, x0 = p - y * 25;
    const float* srow = &Ssh[wave][y * 29 + x0];
    float sum = 0.f;
#pragma unroll
    for (int u = 0; u < 5; ++u)
#pragma unroll
      for (int v = 0; v < 5; ++v)
        sum += srow[u * 29 + v] * kr[u * 5 + v];
    op[(long)p * 256] = (bf16_t)sum;
  }
}

// ------------------------------ fused head ---------------------------------
// h = relu(bn(feat @ W1^T)); out = h @ W2^T + b2.  Block tile 32M x 256N,
// 8 waves in 2(M) x 4(N) grid. A staged once; h round-trips through LDS in
// WMMA A-layout (bf16) between the two GEMMs. Two barriers total.

__global__ __launch_bounds__(256)
void head_fused_wmma(const bf16_t* __restrict__ featb,  // [M][256] bf16
                     const bf16_t* __restrict__ w1b,    // [256][256] bf16
                     const bf16_t* __restrict__ w2b,    // [256][256] bf16
                     const float* __restrict__ scale_h,
                     const float* __restrict__ bias_h,
                     const float* __restrict__ bh2,
                     float* __restrict__ out) {         // NCHW f32, HW=625
  __shared__ __align__(16) bf16_t As[32][264];
  __shared__ __align__(16) bf16_t Hs[32][264];

  const int tid = threadIdx.x, lane = tid & 31, wave = tid >> 5;
  const int waveM = wave & 1, waveN = wave >> 1;
  const long m0 = (long)blockIdx.x * 32;

  // ---- stage entire 32x256 A tile (one barrier) ----
  {
    const int srow = tid >> 3;
    const int skb  = (tid & 7) * 32;
#pragma unroll
    for (int i = 0; i < 4; ++i)
      *(v8bf*)&As[srow][skb + i * 8] =
          *(const v8bf*)&featb[(m0 + srow) * 256 + skb + i * 8];
  }
  __syncthreads();

  const int nLo  = lane & 15;
  const int aRow = waveM * 16 + nLo;
  const int aK   = (lane < 16) ? 0 : 8;
  const int bKH  = (lane < 16) ? 0 : 8;

  // ---- GEMM1 ----
  v8f acc[4] = {};
  for (int k0 = 0; k0 < 256; k0 += 32) {
    Bf16x16 a;
    a.h[0] = *(const v8bf*)&As[aRow][k0 + aK];
    a.h[1] = *(const v8bf*)&As[aRow][k0 + aK + 16];
#pragma unroll
    for (int t = 0; t < 4; ++t) {
      int n = waveN * 64 + t * 16 + nLo;
      const bf16_t* bp = w1b + (long)n * 256 + k0 + bKH;
      Bf16x16 bfr;
      bfr.h[0] = *(const v8bf*)(bp);
      bfr.h[1] = *(const v8bf*)(bp + 16);
      acc[t] = wmma_bf16(a, bfr, acc[t]);
    }
  }

  // ---- BN + ReLU, deposit h as bf16 in LDS (row m, contiguous n) ----
  {
    const int mb = waveM * 16 + ((lane < 16) ? 0 : 8);
#pragma unroll
    for (int t = 0; t < 4; ++t) {
      int n = waveN * 64 + t * 16 + nLo;
      float s = scale_h[n], bi = bias_h[n];
#pragma unroll
      for (int v = 0; v < 8; ++v) {
        float val = acc[t][v] * s + bi;
        val = val > 0.f ? val : 0.f;
        Hs[mb + v][n] = (bf16_t)val;
      }
    }
  }
  __syncthreads();

  // ---- GEMM2 ----
  v8f acc2[4] = {};
  for (int k0 = 0; k0 < 256; k0 += 32) {
    Bf16x16 a;
    a.h[0] = *(const v8bf*)&Hs[aRow][k0 + aK];
    a.h[1] = *(const v8bf*)&Hs[aRow][k0 + aK + 16];
#pragma unroll
    for (int t = 0; t < 4; ++t) {
      int o = waveN * 64 + t * 16 + nLo;
      const bf16_t* bp = w2b + (long)o * 256 + k0 + bKH;
      Bf16x16 bfr;
      bfr.h[0] = *(const v8bf*)(bp);
      bfr.h[1] = *(const v8bf*)(bp + 16);
      acc2[t] = wmma_bf16(a, bfr, acc2[t]);
    }
  }

  // ---- bias + NCHW store ----
  const int mb = waveM * 16 + ((lane < 16) ? 0 : 8);
#pragma unroll
  for (int v = 0; v < 8; ++v) {
    long mg = m0 + mb + v;
    int  bb = (int)(mg / 625);
    int  pp = (int)(mg % 625);
    float* orow = out + (long)bb * 256 * 625 + pp;
#pragma unroll
    for (int t = 0; t < 4; ++t) {
      int o = waveN * 64 + t * 16 + nLo;
      orow[(long)o * 625] = acc2[t][v] + bh2[o];
    }
  }
}

// ------------------------------- launcher ----------------------------------

extern "C" void kernel_launch(void* const* d_in, const int* in_sizes, int n_in,
                              void* d_out, int out_size, void* d_ws, size_t ws_size,
                              hipStream_t stream) {
  const float* kin = (const float*)d_in[0];   // [128,256,7,7]
  const float* sin = (const float*)d_in[1];   // [128,256,31,31]
  const float* wk  = (const float*)d_in[2];
  const float* gk  = (const float*)d_in[3];
  const float* bk  = (const float*)d_in[4];
  const float* mk  = (const float*)d_in[5];
  const float* vk  = (const float*)d_in[6];
  const float* wsw = (const float*)d_in[7];
  const float* gs  = (const float*)d_in[8];
  const float* bs  = (const float*)d_in[9];
  const float* ms  = (const float*)d_in[10];
  const float* vs  = (const float*)d_in[11];
  const float* wh1 = (const float*)d_in[12];
  const float* gh  = (const float*)d_in[13];
  const float* bh  = (const float*)d_in[14];
  const float* mh  = (const float*)d_in[15];
  const float* vh  = (const float*)d_in[16];
  const float* wh2 = (const float*)d_in[17];
  const float* bh2 = (const float*)d_in[18];
  float* out = (float*)d_out;

  // workspace carve-up (256B aligned)
  uint8_t* base = (uint8_t*)d_ws;
  size_t off = 0;
  auto carve = [&](size_t bytes) -> void* {
    void* p = base + off;
    off = (off + bytes + 255) & ~(size_t)255;
    return p;
  };
  bf16_t* wkb   = (bf16_t*)carve(589824ull * 2);        // [256][2304]
  bf16_t* wsb   = (bf16_t*)carve(589824ull * 2);
  bf16_t* w1b   = (bf16_t*)carve(65536ull * 2);         // [256][256]
  bf16_t* w2b   = (bf16_t*)carve(65536ull * 2);
  float*  sck   = (float*)carve(256 * 4);
  float*  bik   = (float*)carve(256 * 4);
  float*  scs   = (float*)carve(256 * 4);
  float*  bis   = (float*)carve(256 * 4);
  float*  sch   = (float*)carve(256 * 4);
  float*  bih   = (float*)carve(256 * 4);
  float*  kfeat = (float*)carve(819200ull * 4);         // [128,256,5,5]
  float*  sfeat = (float*)carve(27557888ull * 4);       // [128,256,29,29]
  bf16_t* featb = (bf16_t*)carve(20480000ull * 2);      // [128*625][256]

  // weight conversion + BN folding
  cvt_f32_to_bf16<<<(589824 + 255) / 256, 256, 0, stream>>>(wk,  wkb, 589824);
  cvt_f32_to_bf16<<<(589824 + 255) / 256, 256, 0, stream>>>(wsw, wsb, 589824);
  cvt_f32_to_bf16<<<(65536 + 255) / 256, 256, 0, stream>>>(wh1, w1b, 65536);
  cvt_f32_to_bf16<<<(65536 + 255) / 256, 256, 0, stream>>>(wh2, w2b, 65536);
  bn_fold<<<1, 256, 0, stream>>>(gk, bk, mk, vk, sck, bik);
  bn_fold<<<1, 256, 0, stream>>>(gs, bs, ms, vs, scs, bis);
  bn_fold<<<1, 256, 0, stream>>>(gh, bh, mh, vh, sch, bih);

  // conv towers (implicit GEMM, WMMA). Full N=256 per block.
  // kernel tower: M = 128*25 = 3200 -> 50 M-blocks
  conv3x3_bnrelu_wmma<<<dim3(50, 1), 256, 0, stream>>>(
      kin, wkb, sck, bik, kfeat, 7, 7, 5, 5);
  // search tower: M = 128*841 = 107648 -> 1682 M-blocks
  conv3x3_bnrelu_wmma<<<dim3(1682, 1), 256, 0, stream>>>(
      sin, wsb, scs, bis, sfeat, 31, 31, 29, 29);

  // depthwise cross-correlation: 32768 (b,c) pairs, 8 waves/block
  xcorr_dw<<<4096, 256, 0, stream>>>(sfeat, kfeat, featb);

  // fused head: M = 128*625 = 80000 -> 2500 M-blocks
  head_fused_wmma<<<2500, 256, 0, stream>>>(featb, w1b, w2b, sch, bih, bh2, out);
}
